// DilatedAttention_73332271612511
// MI455X (gfx1250) — matrix operand order repaired and checked by reading, not compiled, not run on hardware
//
#include <hip/hip_runtime.h>

typedef __attribute__((ext_vector_type(16))) __bf16 v16bf;
typedef __attribute__((ext_vector_type(8)))  float  v8f;

static constexpr int Bb = 4, Cc = 512, Hh = 128, Ww = 128, CQK = 64;
static constexpr int HWc = Hh * Ww;

__device__ inline unsigned short f2bf_bits(float f) {
    unsigned int x = __float_as_uint(f);
    unsigned int lsb = (x >> 16) & 1u;
    x += 0x7fffu + lsb;              // round-to-nearest-even
    return (unsigned short)(x >> 16);
}

union FragU {
    uint4 u[2];
    v16bf v;
    unsigned short s[16];
};

// 16 contiguous bf16 starting at p (16B aligned): B-fragment (lanes 0-15 K 0..15,
// lanes 16-31 K 16..31 selected by caller via +kh16)
__device__ inline v16bf ldfrag_contig(const unsigned short* p) {
    FragU f;
    f.u[0] = *(const uint4*)p;
    f.u[1] = *(const uint4*)(p + 8);
    return f.v;
}
// A-fragment: lanes 0-15 take K [0,16), lanes 16-31 take K [16,32): two aligned 16B loads
__device__ inline v16bf ldfrag_a(const unsigned short* p, int kh8) {
    FragU f;
    f.u[0] = *(const uint4*)(p + 2 * kh8);
    f.u[1] = *(const uint4*)(p + 2 * kh8 + 8);
    return f.v;
}

// ---------------- weights -> bf16 ----------------
__global__ void convert_w(const float* __restrict__ Wq, const float* __restrict__ Wk,
                          const float* __restrict__ Wv, unsigned short* __restrict__ wb) {
    const int NWQ = CQK * Cc;
    const int NW  = 2 * NWQ + Cc * Cc;
    for (int i = blockIdx.x * blockDim.x + threadIdx.x; i < NW;
         i += gridDim.x * blockDim.x) {
        float v;
        if (i < NWQ)            v = Wq[i];
        else if (i < 2 * NWQ)   v = Wk[i - NWQ];
        else                    v = Wv[i - 2 * NWQ];
        wb[i] = f2bf_bits(v);
    }
}

// ---------------- x (B,C,HW) f32 -> x_T (B,HW,C) bf16, tiled transpose ----------------
__global__ void xpose_x(const float* __restrict__ x, unsigned short* __restrict__ xT) {
    __shared__ unsigned short tile[32][33];
    const int b   = blockIdx.z;
    const int hw0 = blockIdx.x * 32;
    const int c0  = blockIdx.y * 32;
    const int tx = threadIdx.x, ty = threadIdx.y;        // 32 x 8
    const float* xb = x + (size_t)b * Cc * HWc;
    unsigned short* xtb = xT + (size_t)b * HWc * Cc;
#pragma unroll
    for (int r = 0; r < 4; ++r)
        tile[ty + r * 8][tx] = f2bf_bits(xb[(size_t)(c0 + ty + r * 8) * HWc + hw0 + tx]);
    __syncthreads();
#pragma unroll
    for (int r = 0; r < 4; ++r)
        xtb[(size_t)(hw0 + ty + r * 8) * Cc + c0 + tx] = tile[tx][ty + r * 8];
}

// ---------------- v_std (B,C,H,W) -> v_swap (B,C,W,H), per-plane transpose ----------------
__global__ void vswap_k(const unsigned short* __restrict__ vstd,
                        unsigned short* __restrict__ vsw) {
    __shared__ unsigned short tile[32][33];
    const int plane = blockIdx.z;                        // b*Cc + c
    const int h0 = blockIdx.x * 32, w0 = blockIdx.y * 32;
    const int tx = threadIdx.x, ty = threadIdx.y;
    const unsigned short* p = vstd + (size_t)plane * HWc;
    unsigned short* q = vsw + (size_t)plane * HWc;
#pragma unroll
    for (int r = 0; r < 4; ++r)
        tile[ty + r * 8][tx] = p[(size_t)(h0 + ty + r * 8) * Ww + w0 + tx];
    __syncthreads();
#pragma unroll
    for (int r = 0; r < 4; ++r)
        q[(size_t)(w0 + ty + r * 8) * Hh + h0 + tx] = tile[tx][ty + r * 8];
}

// ---------------- q/k projection: out_T[b][hw][o] = sum_c W[o,c]*x_T[b][hw][c] + bias
// All operands contiguous in K; packed 16B epilogue store (8 consecutive o per lane).
__global__ void proj_qk(const unsigned short* __restrict__ xT,
                        const unsigned short* __restrict__ Wbf,
                        const float* __restrict__ bias,
                        unsigned short* __restrict__ outT, int O) {
    const int tid = threadIdx.x;
    const int wave = tid >> 5, lane = tid & 31;
    const int n  = (blockIdx.x * 8 + wave) * 16 + (lane & 15);
    const int m0 = blockIdx.y * 16;
    const int b  = blockIdx.z;
    const int mrow = lane & 15;
    const int kh8 = (lane >> 4) * 8, kh16 = (lane >> 4) * 16, hi = lane >> 4;
    const unsigned short* pa = Wbf + (size_t)(m0 + mrow) * Cc;
    const unsigned short* pb = xT + ((size_t)b * HWc + n) * Cc;
    v8f acc = {};
    for (int kk = 0; kk < Cc; kk += 32) {
        if (kk + 32 < Cc) __builtin_prefetch(pb + kk + 32, 0, 1);
        v16bf a  = ldfrag_a(pa + kk, kh8);
        v16bf bb = ldfrag_contig(pb + kk + kh16);
        acc = __builtin_amdgcn_wmma_f32_16x16x32_bf16(false, a, false, bb,
                                                      (short)0, acc, false, false);
    }
    FragU f;
#pragma unroll
    for (int r = 0; r < 8; ++r) f.s[r] = f2bf_bits(acc[r] + bias[m0 + hi * 8 + r]);
    *(uint4*)(outT + ((size_t)b * HWc + n) * O + m0 + hi * 8) = f.u[0];
}

// ---------------- v projection: v_std[b][o][hw] (standard layout epilogue) ----------------
__global__ void proj_v(const unsigned short* __restrict__ xT,
                       const unsigned short* __restrict__ Wbf,
                       const float* __restrict__ bias,
                       unsigned short* __restrict__ vstd) {
    const int tid = threadIdx.x;
    const int wave = tid >> 5, lane = tid & 31;
    const int n  = (blockIdx.x * 8 + wave) * 16 + (lane & 15);
    const int m0 = blockIdx.y * 16;
    const int b  = blockIdx.z;
    const int mrow = lane & 15;
    const int kh8 = (lane >> 4) * 8, kh16 = (lane >> 4) * 16, hi = lane >> 4;
    const unsigned short* pa = Wbf + (size_t)(m0 + mrow) * Cc;
    const unsigned short* pb = xT + ((size_t)b * HWc + n) * Cc;
    v8f acc = {};
    for (int kk = 0; kk < Cc; kk += 32) {
        if (kk + 32 < Cc) __builtin_prefetch(pb + kk + 32, 0, 1);
        v16bf a  = ldfrag_a(pa + kk, kh8);
        v16bf bb = ldfrag_contig(pb + kk + kh16);
        acc = __builtin_amdgcn_wmma_f32_16x16x32_bf16(false, a, false, bb,
                                                      (short)0, acc, false, false);
    }
    unsigned short* ob = vstd + (size_t)b * Cc * HWc;
#pragma unroll
    for (int r = 0; r < 8; ++r) {
        int m = m0 + r + hi * 8;
        ob[(size_t)m * HWc + n] = f2bf_bits(acc[r] + bias[m]);
    }
}

// ---------------- row scores: E[w,i] = sum_c q[c,w]*k[c,i] per (b,h); direct-global frags
__global__ void escore_row(const unsigned short* __restrict__ qT,
                           const unsigned short* __restrict__ kT,
                           float* __restrict__ scores) {
    const int h = blockIdx.x, b = blockIdx.y;
    const int tid = threadIdx.x;
    const int wave = tid >> 5, lane = tid & 31;
    const int m0 = wave * 16;
    const int mrow = lane & 15, ncol = lane & 15;
    const int kh8 = (lane >> 4) * 8, kh16 = (lane >> 4) * 16, hi = lane >> 4;
    const unsigned short* qrow = qT + ((size_t)b * HWc + (size_t)h * Ww) * CQK;
    const unsigned short* krow = kT + ((size_t)b * HWc + (size_t)h * Ww) * CQK;
    v16bf a2[2];
#pragma unroll
    for (int kq = 0; kq < 2; ++kq)
        a2[kq] = ldfrag_a(qrow + (size_t)(m0 + mrow) * CQK + kq * 32, kh8);
    float* srow = scores + ((size_t)b * Hh + h) * Ww * 256;
    for (int nt = 0; nt < 8; ++nt) {
        const int n0 = nt * 16;
        v8f acc = {};
#pragma unroll
        for (int kq = 0; kq < 2; ++kq) {
            v16bf bb = ldfrag_contig(krow + (size_t)(n0 + ncol) * CQK + kq * 32 + kh16);
            acc = __builtin_amdgcn_wmma_f32_16x16x32_bf16(false, a2[kq], false, bb,
                                                          (short)0, acc, false, false);
        }
#pragma unroll
        for (int r = 0; r < 8; ++r) {
            int m = m0 + r + hi * 8;             // w index
            srow[(size_t)m * 256 + n0 + ncol] = acc[r];
        }
    }
}

// ---------------- col scores: E[h,j] = sum_c q[c,h]*k[c,j] per (b,w), diag -inf
__global__ void escore_col(const unsigned short* __restrict__ qT,
                           const unsigned short* __restrict__ kT,
                           float* __restrict__ scores) {
    const int w = blockIdx.x, b = blockIdx.y;
    const int tid = threadIdx.x;
    const int wave = tid >> 5, lane = tid & 31;
    const int m0 = wave * 16;
    const int mrow = lane & 15, ncol = lane & 15;
    const int kh8 = (lane >> 4) * 8, kh16 = (lane >> 4) * 16, hi = lane >> 4;
    const size_t rstride = (size_t)Ww * CQK;     // pixel-row stride along h
    const unsigned short* qcol = qT + ((size_t)b * HWc + w) * CQK;
    const unsigned short* kcol = kT + ((size_t)b * HWc + w) * CQK;
    v16bf a2[2];
#pragma unroll
    for (int kq = 0; kq < 2; ++kq)
        a2[kq] = ldfrag_a(qcol + (size_t)(m0 + mrow) * rstride + kq * 32, kh8);
    for (int nt = 0; nt < 8; ++nt) {
        const int n0 = nt * 16;
        v8f acc = {};
#pragma unroll
        for (int kq = 0; kq < 2; ++kq) {
            v16bf bb = ldfrag_contig(kcol + (size_t)(n0 + ncol) * rstride + kq * 32 + kh16);
            acc = __builtin_amdgcn_wmma_f32_16x16x32_bf16(false, a2[kq], false, bb,
                                                          (short)0, acc, false, false);
        }
#pragma unroll
        for (int r = 0; r < 8; ++r) {
            int m = m0 + r + hi * 8;             // h index
            int n = n0 + ncol;                   // j index
            float val = (m == n) ? -__builtin_inff() : acc[r];
            scores[(((size_t)b * Hh + m) * Ww + w) * 256 + 128 + n] = val;
        }
    }
}

// ---------------- softmax over 256 scores per pixel, emit bf16
__global__ void softmax256(const float* __restrict__ scores,
                           unsigned short* __restrict__ abf) {
    __shared__ float red[256];
    const int w = blockIdx.x, h = blockIdx.y, b = blockIdx.z;
    const size_t base = (((size_t)b * Hh + h) * Ww + w) * 256;
    const int t = threadIdx.x;
    float v = scores[base + t];
    red[t] = v;
    __syncthreads();
    for (int s = 128; s > 0; s >>= 1) {
        if (t < s) red[t] = fmaxf(red[t], red[t + s]);
        __syncthreads();
    }
    float mx = red[0];
    __syncthreads();
    float e = __expf(v - mx);
    red[t] = e;
    __syncthreads();
    for (int s = 128; s > 0; s >>= 1) {
        if (t < s) red[t] += red[t + s];
        __syncthreads();
    }
    abf[base + t] = f2bf_bits(e / red[0]);
}

// ---------------- out_row: out[b,c,h,w] = sum_i v[b,c,h,i]*a_row[b,i,h,w]
__global__ void out_row(const unsigned short* __restrict__ vstd,
                        const unsigned short* __restrict__ abf,
                        float* __restrict__ out) {
    const int h = blockIdx.x, b = blockIdx.y;
    const int tid = threadIdx.x;
    const int wave = tid >> 5, lane = tid & 31;
    const int mrow = lane & 15, ncol = lane & 15;
    const int kh8 = (lane >> 4) * 8, kh16 = (lane >> 4) * 16, hi = lane >> 4;
    const unsigned short* vbb = vstd + (size_t)b * Cc * HWc + (size_t)h * Ww;
    const unsigned short* arow = abf + ((size_t)b * Hh + h) * Ww * 256;
    float* ob = out + (size_t)b * Cc * HWc + (size_t)h * Ww;
    for (int mt = wave; mt < Cc / 16; mt += 8) {
        const int m0 = mt * 16;
        const unsigned short* pa = vbb + (size_t)(m0 + mrow) * HWc;
        v16bf a4[4];
#pragma unroll
        for (int kq = 0; kq < 4; ++kq) a4[kq] = ldfrag_a(pa + kq * 32, kh8);
        for (int nt = 0; nt < 8; ++nt) {
            const int n0 = nt * 16;
            v8f acc = {};
#pragma unroll
            for (int kq = 0; kq < 4; ++kq) {
                v16bf bb = ldfrag_contig(arow + (size_t)(n0 + ncol) * 256 + kq * 32 + kh16);
                acc = __builtin_amdgcn_wmma_f32_16x16x32_bf16(false, a4[kq], false, bb,
                                                              (short)0, acc, false, false);
            }
#pragma unroll
            for (int r = 0; r < 8; ++r) {
                int m = m0 + r + hi * 8;
                ob[(size_t)m * HWc + n0 + ncol] = acc[r];
            }
        }
    }
}

// ---------------- out_col: out[b,c,h,w] += sum_j v[b,c,j,w]*a_col[b,j,h,w]
__global__ void out_col(const unsigned short* __restrict__ vsw,
                        const unsigned short* __restrict__ abf,
                        float* __restrict__ out) {
    const int w = blockIdx.x, b = blockIdx.y;
    const int tid = threadIdx.x;
    const int wave = tid >> 5, lane = tid & 31;
    const int mrow = lane & 15, ncol = lane & 15;
    const int kh8 = (lane >> 4) * 8, kh16 = (lane >> 4) * 16, hi = lane >> 4;
    const unsigned short* vbb = vsw + (size_t)b * Cc * HWc + (size_t)w * Hh;  // [c][h]
    float* ob = out + (size_t)b * Cc * HWc + w;
    for (int mt = wave; mt < Cc / 16; mt += 8) {
        const int m0 = mt * 16;
        const unsigned short* pa = vbb + (size_t)(m0 + mrow) * HWc;
        v16bf a4[4];
#pragma unroll
        for (int kq = 0; kq < 4; ++kq) a4[kq] = ldfrag_a(pa + kq * 32, kh8);
        for (int nt = 0; nt < 8; ++nt) {
            const int n0 = nt * 16;
            v8f acc = {};
#pragma unroll
            for (int kq = 0; kq < 4; ++kq) {
                v16bf bb = ldfrag_contig(abf + (((size_t)b * Hh + n0 + ncol) * Ww + w) * 256
                                             + 128 + kq * 32 + kh16);
                acc = __builtin_amdgcn_wmma_f32_16x16x32_bf16(false, a4[kq], false, bb,
                                                              (short)0, acc, false, false);
            }
#pragma unroll
            for (int r = 0; r < 8; ++r) {
                int m = m0 + r + hi * 8;
                ob[(size_t)m * HWc + (size_t)(n0 + ncol) * Ww] += acc[r];
            }
        }
    }
}

extern "C" void kernel_launch(void* const* d_in, const int* in_sizes, int n_in,
                              void* d_out, int out_size, void* d_ws, size_t ws_size,
                              hipStream_t stream) {
    const float* x  = (const float*)d_in[0];
    const float* Wq = (const float*)d_in[1];
    const float* bq = (const float*)d_in[2];
    const float* Wk = (const float*)d_in[3];
    const float* bk = (const float*)d_in[4];
    const float* Wv = (const float*)d_in[5];
    const float* bv = (const float*)d_in[6];
    float* out = (float*)d_out;

    char* ws = (char*)d_ws;
    size_t off = 0;
    unsigned short* xT   = (unsigned short*)(ws + off); off += (size_t)Bb * Cc * HWc * 2;
    unsigned short* qTb  = (unsigned short*)(ws + off); off += (size_t)Bb * CQK * HWc * 2;
    unsigned short* kTb  = (unsigned short*)(ws + off); off += (size_t)Bb * CQK * HWc * 2;
    unsigned short* vstd = (unsigned short*)(ws + off); off += (size_t)Bb * Cc * HWc * 2;
    unsigned short* vsw  = (unsigned short*)(ws + off); off += (size_t)Bb * Cc * HWc * 2;
    unsigned short* wb   = (unsigned short*)(ws + off); off += (size_t)(2 * CQK * Cc + Cc * Cc) * 2;
    float* scores = (float*)(ws + off);                 off += (size_t)Bb * Hh * Ww * 256 * 4;
    unsigned short* abf = (unsigned short*)(ws + off);

    convert_w<<<1280, 256, 0, stream>>>(Wq, Wk, Wv, wb);
    xpose_x<<<dim3(HWc / 32, Cc / 32, Bb), dim3(32, 8), 0, stream>>>(x, xT);
    proj_qk<<<dim3(HWc / 128, CQK / 16, Bb), 256, 0, stream>>>(xT, wb, bq, qTb, CQK);
    proj_qk<<<dim3(HWc / 128, CQK / 16, Bb), 256, 0, stream>>>(xT, wb + (size_t)CQK * Cc, bk, kTb, CQK);
    proj_v<<<dim3(HWc / 128, Cc / 16, Bb), 256, 0, stream>>>(xT, wb + (size_t)2 * CQK * Cc, bv, vstd);
    vswap_k<<<dim3(Hh / 32, Ww / 32, Bb * Cc), dim3(32, 8), 0, stream>>>(vstd, vsw);
    escore_row<<<dim3(Hh, Bb), 256, 0, stream>>>(qTb, kTb, scores);
    escore_col<<<dim3(Ww, Bb), 256, 0, stream>>>(qTb, kTb, scores);
    softmax256<<<dim3(Ww, Hh, Bb), 256, 0, stream>>>(scores, abf);
    out_row<<<dim3(Hh, Bb), 256, 0, stream>>>(vstd, abf, out);
    out_col<<<dim3(Ww, Bb), 256, 0, stream>>>(vsw, abf, out);
}
